// BertLSTMCrfModel_31224412241924
// MI455X (gfx1250) — matrix operand order repaired and checked by reading, not compiled, not run on hardware
//
#include <hip/hip_runtime.h>
#include <hip/hip_bf16.h>
#include <math.h>

typedef unsigned short u16;
typedef unsigned int   u32;
typedef unsigned long long u64;
typedef __attribute__((ext_vector_type(4)))  unsigned int u32x4;
typedef __attribute__((ext_vector_type(16))) __bf16       v16bf;
typedef __attribute__((ext_vector_type(8)))  float        v8f;

union FragAB { v16bf v; u32x4 q[2]; };

#define BB 64
#define SS 512
#define DD 768
#define HH 384
#define G4 1536                 // 4*H
#define TT 9
#define MROWS (BB * SS)         // 32768

__device__ __forceinline__ u16 f2bf(float f) {
  u32 u = __float_as_uint(f);
  u32 r = (u >> 16) & 1u;
  u += 0x7fffu + r;
  return (u16)(u >> 16);
}
__device__ __forceinline__ float bf2f(u16 h) {
  return __uint_as_float(((u32)h) << 16);
}
__device__ __forceinline__ float sigmf(float x) { return 1.0f / (1.0f + expf(-x)); }

// ---------------------------------------------------------------- convert x -> bf16
__global__ __launch_bounds__(256) void k_cvt(const float* __restrict__ x,
                                             u16* __restrict__ xb, int n) {
  int i = blockIdx.x * 256 + threadIdx.x;
  if (i < n) xb[i] = f2bf(x[i]);
}

// ------------------------------------------- pack W (Ndim x Kdim row-major, fp32)
// into WMMA B-fragment order: dst[((kt*NT+nt)*32+lane)*16+e] = W[N][K], bf16
// K = kt*32 + (lane>>4)*16 + e ; N = nt*16 + (lane&15)
__global__ __launch_bounds__(256) void k_pack(const float* __restrict__ W,
                                              u16* __restrict__ dst,
                                              int Kdim, int NT, int total) {
  int id = blockIdx.x * 256 + threadIdx.x;
  if (id >= total) return;
  int e = id & 15;
  int lane = (id >> 4) & 31;
  int rest = id >> 9;
  int nt = rest % NT;
  int kt = rest / NT;
  int K = kt * 32 + (lane >> 4) * 16 + e;
  int N = nt * 16 + (lane & 15);
  dst[id] = f2bf(W[(size_t)N * Kdim + K]);
}

// ---------------------------------------------------------------- bias = bih + bhh
__global__ __launch_bounds__(256) void k_bias(const float* bf1, const float* bf2,
                                              const float* bb1, const float* bb2,
                                              float* __restrict__ bias) {
  int i = blockIdx.x * 256 + threadIdx.x;
  if (i < G4) bias[i] = bf1[i] + bf2[i];
  else if (i < 2 * G4) bias[i] = bb1[i - G4] + bb2[i - G4];
}

// --------------------------------- xg = x @ Wih^T + bias   (both directions)
// grid = 2(dir) * 24(Nblk of 64) * 256(Mblk of 128), block = 256 (8 waves)
// B tile (4 KB per k-step) staged into LDS once per WG via async-to-LDS,
// double-buffered; A fragments software-pipelined in registers.
__global__ __launch_bounds__(256) void k_gemm_xg(const u16* __restrict__ xb,
                                                 const u16* __restrict__ wihp,
                                                 const float* __restrict__ bias,
                                                 float* __restrict__ xg) {
  __shared__ __align__(16) u16 sb[2][2048];   // 2 x 4KB B-tile buffers

  int bi = blockIdx.x;
  int dir = bi / (24 * 256);
  int rem = bi % (24 * 256);
  int nb = rem / 256;
  int mb = rem % 256;

  const u16* wp = wihp + (size_t)dir * 24 * 96 * 512;
  const float* bs = bias + dir * G4;
  float* out = xg + (size_t)dir * MROWS * G4;

  int tid = threadIdx.x;
  int wv = tid >> 5;
  int lane = tid & 31;
  int nl = lane & 15;
  int ml = (lane < 16) ? 0 : 8;
  int rowA = mb * 128 + wv * 16 + (lane & 15);
  int kA = (lane < 16) ? 0 : 16;

  v8f acc[4];
  for (int nt = 0; nt < 4; ++nt) {
    float bv = bs[nb * 64 + nt * 16 + nl];
    for (int r = 0; r < 8; ++r) acc[nt][r] = bv;
  }

  FragAB a[2];
  auto loadA = [&](int s, int k) {
    const u16* ap = xb + (size_t)rowA * DD + k * 32 + kA;
    a[s].q[0] = *(const u32x4*)(ap);
    a[s].q[1] = *(const u32x4*)(ap + 8);
  };
  // async-copy the whole 4 KB B tile for k-step into LDS buffer s:
  // 256 threads x 16 B each; source is the packed fragment stream (contiguous).
  auto issueB = [&](int s, int k) {
    const u16* src = wp + (size_t)(k * 96 + nb * 4) * 512 + tid * 8;
    u32 dst = (u32)(size_t)(&sb[s][tid * 8]);       // flat low 32 bits = LDS offset
    asm volatile("global_load_async_to_lds_b128 %0, %1, off"
                 :: "v"(dst), "v"((u64)(size_t)src)
                 : "memory");
  };

  issueB(0, 0);
  loadA(0, 0);
#pragma unroll
  for (int k = 0; k < 24; ++k) {
    int cur = k & 1;
    int nxt = 1 - cur;
    if (k + 1 < 24) {               // issue next tile before consuming current
      issueB(nxt, k + 1);
      loadA(nxt, k + 1);
      asm volatile("s_wait_asynccnt 0x1" ::: "memory");
    } else {
      asm volatile("s_wait_asynccnt 0x0" ::: "memory");
    }
    __syncthreads();                // current tile resident in LDS for all waves
#pragma unroll
    for (int nt = 0; nt < 4; ++nt) {
      const u16* bp = &sb[cur][nt * 512 + lane * 16];
      FragAB b;
      b.q[0] = *(const u32x4*)(bp);
      b.q[1] = *(const u32x4*)(bp + 8);
      acc[nt] = __builtin_amdgcn_wmma_f32_16x16x32_bf16(
          false, a[cur].v, false, b.v, (short)0, acc[nt], false, false);
    }
    __syncthreads();                // all reads done before buffer is re-filled
  }

  for (int nt = 0; nt < 4; ++nt)
    for (int r = 0; r < 8; ++r)
      out[(size_t)(mb * 128 + wv * 16 + ml + r) * G4 + nb * 64 + nt * 16 + nl] =
          acc[nt][r];
}

// ---------------------------------------------------------------- grid barrier
__device__ __forceinline__ void gbar(int* cnt, int* gen, int nwg) {
  __syncthreads();
  if (threadIdx.x == 0) {
    __builtin_amdgcn_fence(__ATOMIC_RELEASE, "agent");
    int g = __hip_atomic_load(gen, __ATOMIC_RELAXED, __HIP_MEMORY_SCOPE_AGENT);
    int a = __hip_atomic_fetch_add(cnt, 1, __ATOMIC_ACQ_REL, __HIP_MEMORY_SCOPE_AGENT);
    if (a == nwg - 1) {
      __hip_atomic_store(cnt, 0, __ATOMIC_RELAXED, __HIP_MEMORY_SCOPE_AGENT);
      __hip_atomic_fetch_add(gen, 1, __ATOMIC_RELEASE, __HIP_MEMORY_SCOPE_AGENT);
    } else {
      while (__hip_atomic_load(gen, __ATOMIC_RELAXED, __HIP_MEMORY_SCOPE_AGENT) == g)
        __builtin_amdgcn_s_sleep(2);
    }
    __builtin_amdgcn_fence(__ATOMIC_ACQUIRE, "agent");
  }
  __syncthreads();
}

// --------------------------------- persistent bidirectional LSTM recurrence
// grid = 48 (24 fwd + 24 bwd), block = 128 (4 waves).
// WG owns 16 hidden cols (all 4 gates); wave owns a 16-batch stripe.
__global__ __launch_bounds__(128) void k_lstm(const float* __restrict__ xg,
                                              const u16* __restrict__ whhp,
                                              u16* __restrict__ hst,
                                              u16* __restrict__ hcat,
                                              int* __restrict__ bar_cnt,
                                              int* __restrict__ bar_gen) {
  int dir = blockIdx.x / 24;
  int wg = blockIdx.x % 24;
  int wv = threadIdx.x >> 5;
  int lane = threadIdx.x & 31;
  int nl = lane & 15;
  int ml = (lane < 16) ? 0 : 8;
  int rowA = wv * 16 + (lane & 15);
  int kA = (lane < 16) ? 0 : 16;

  const float* xg_d = xg + (size_t)dir * MROWS * G4;
  const u16* wp = whhp + (size_t)dir * 12 * 96 * 512;
  u16* h0 = hst + (size_t)dir * 2 * BB * HH;
  int* cnt = bar_cnt + dir;
  int* gen = bar_gen + dir;

  float c[8];
  for (int r = 0; r < 8; ++r) c[r] = 0.0f;

  for (int t = 0; t < SS; ++t) {
    int p = t & 1;
    const u16* hrd = h0 + (size_t)p * BB * HH;
    u16* hwr = h0 + (size_t)(1 - p) * BB * HH;
    int spos = dir ? (SS - 1 - t) : t;

    v8f acc[4];
#pragma unroll
    for (int g = 0; g < 4; ++g) {
      int col = g * HH + wg * 16 + nl;
      for (int r = 0; r < 8; ++r) {
        int brow = wv * 16 + ml + r;
        acc[g][r] = xg_d[(size_t)(brow * SS + spos) * G4 + col];
      }
    }

    FragAB a[2];
    FragAB bfr[2][4];
    auto loadA = [&](int s, int k) {
      const u16* ap = hrd + (size_t)rowA * HH + k * 32 + kA;
      a[s].q[0] = *(const u32x4*)(ap);
      a[s].q[1] = *(const u32x4*)(ap + 8);
    };
    auto loadB = [&](int s, int k) {
#pragma unroll
      for (int g = 0; g < 4; ++g) {
        const u16* bp = wp + ((size_t)(k * 96 + g * 24 + wg) * 32 + lane) * 16;
        bfr[s][g].q[0] = *(const u32x4*)(bp);
        bfr[s][g].q[1] = *(const u32x4*)(bp + 8);
      }
    };

    loadA(0, 0);
    loadB(0, 0);
#pragma unroll
    for (int k = 0; k < 12; ++k) {
      int cur = k & 1;
      int nxt = 1 - cur;
      if (k + 1 < 12) {
        loadA(nxt, k + 1);
        loadB(nxt, k + 1);
      }
#pragma unroll
      for (int g = 0; g < 4; ++g)
        acc[g] = __builtin_amdgcn_wmma_f32_16x16x32_bf16(
            false, a[cur].v, false, bfr[cur][g].v, (short)0, acc[g], false, false);
    }

    for (int r = 0; r < 8; ++r) {
      float iv = sigmf(acc[0][r]);
      float fv = sigmf(acc[1][r]);
      float gv = tanhf(acc[2][r]);
      float ov = sigmf(acc[3][r]);
      float cv = fv * c[r] + iv * gv;
      c[r] = cv;
      float hv = ov * tanhf(cv);
      int brow = wv * 16 + ml + r;
      int hcol = wg * 16 + nl;
      u16 hb = f2bf(hv);
      hwr[(size_t)brow * HH + hcol] = hb;
      hcat[(size_t)(brow * SS + spos) * (2 * HH) + dir * HH + hcol] = hb;
    }

    // prefetch next step's gate pre-activations across the barrier wait
    if (t + 1 < SS) {
      int spn = dir ? (SS - 2 - t) : (t + 1);
#pragma unroll
      for (int g = 0; g < 4; ++g) {
        int col = g * HH + wg * 16 + nl;
        __builtin_prefetch(&xg_d[(size_t)((wv * 16 + ml) * SS + spn) * G4 + col], 0, 3);
      }
    }

    gbar(cnt, gen, 24);
  }
}

// --------------------------------- collapse MLP: tmp1 = W2@W1 [128,768]
__global__ __launch_bounds__(256) void k_tmp1(const float* __restrict__ W2,
                                              const float* __restrict__ W1,
                                              float* __restrict__ tmp1) {
  int id = blockIdx.x * 256 + threadIdx.x;
  if (id >= 128 * 768) return;
  int i = id / 768, j = id % 768;
  float s = 0.0f;
  for (int k = 0; k < 256; ++k) s += W2[i * 256 + k] * W1[k * 768 + j];
  tmp1[id] = s;
}

// Weff = Wc @ tmp1  [9,768]
__global__ __launch_bounds__(256) void k_weff(const float* __restrict__ Wc,
                                              const float* __restrict__ tmp1,
                                              float* __restrict__ weff) {
  int id = blockIdx.x * 256 + threadIdx.x;
  if (id >= TT * 768) return;
  int tq = id / 768, j = id % 768;
  float s = 0.0f;
  for (int k = 0; k < 128; ++k) s += Wc[tq * 128 + k] * tmp1[k * 768 + j];
  weff[id] = s;
}

// beff = bc + Wc @ (b2 + W2 @ b1)
__global__ __launch_bounds__(128) void k_beff(const float* W2, const float* b1,
                                              const float* b2, const float* Wc,
                                              const float* bc,
                                              float* __restrict__ beff) {
  __shared__ float t2[128];
  int tid = threadIdx.x;
  float s = b2[tid];
  for (int k = 0; k < 256; ++k) s += W2[tid * 256 + k] * b1[k];
  t2[tid] = s;
  __syncthreads();
  if (tid < TT) {
    float e = bc[tid];
    for (int k = 0; k < 128; ++k) e += Wc[tid * 128 + k] * t2[k];
    beff[tid] = e;
  }
}

// emissions: em[row][tag] = beff[tag] + hcat_row . weff_tag
__global__ __launch_bounds__(256) void k_emis(const u16* __restrict__ hcat,
                                              const float* __restrict__ weff,
                                              const float* __restrict__ beff,
                                              float* __restrict__ em) {
  int id = blockIdx.x * 256 + threadIdx.x;
  if (id >= MROWS * TT) return;
  int row = id / TT, tag = id - row * TT;
  const u16* hp = hcat + (size_t)row * (2 * HH);
  const float* wp = weff + tag * (2 * HH);
  float s = beff[tag];
#pragma unroll 8
  for (int j = 0; j < 2 * HH; ++j) s += bf2f(hp[j]) * wp[j];
  em[id] = s;
}

// --------------------------------- Viterbi: one wave per batch
__global__ __launch_bounds__(32) void k_vit(const float* __restrict__ em,
                                            const unsigned char* __restrict__ mask,
                                            const float* __restrict__ start,
                                            const float* __restrict__ endt,
                                            const float* __restrict__ trans,
                                            int* __restrict__ hist,
                                            int* __restrict__ out) {
  __shared__ float s_sc[TT];
  __shared__ float s_tr[TT * TT];
  __shared__ float s_new[TT];
  __shared__ int s_idx[TT];
  int b = blockIdx.x;
  int lane = threadIdx.x;
  for (int i = lane; i < TT * TT; i += 32) s_tr[i] = trans[i];
  if (lane < TT) s_sc[lane] = start[lane] + em[(size_t)(b * SS) * TT + lane];
  __syncthreads();

  for (int t = 1; t < SS; ++t) {
    if (lane < TT) {
      float best = -3.4e38f;
      int bi = 0;
      for (int i = 0; i < TT; ++i) {
        float v = s_sc[i] + s_tr[i * TT + lane];
        if (v > best) { best = v; bi = i; }
      }
      s_new[lane] = best + em[(size_t)(b * SS + t) * TT + lane];
      s_idx[lane] = bi;
    }
    __syncthreads();
    if (lane < TT) {
      hist[((size_t)b * (SS - 1) + (t - 1)) * TT + lane] = s_idx[lane];
      if (mask[b * SS + t]) s_sc[lane] = s_new[lane];
    }
    __syncthreads();
  }

  if (lane == 0) {
    float best = -3.4e38f;
    int last = 0;
    for (int j = 0; j < TT; ++j) {
      float v = s_sc[j] + endt[j];
      if (v > best) { best = v; last = j; }
    }
    out[b * SS + SS - 1] = last;
    int tag = last;
    for (int t = SS - 2; t >= 0; --t) {
      if (mask[b * SS + t + 1])
        tag = hist[((size_t)b * (SS - 1) + t) * TT + tag];
      out[b * SS + t] = tag;
    }
  }
}

// =====================================================================
extern "C" void kernel_launch(void* const* d_in, const int* in_sizes, int n_in,
                              void* d_out, int out_size, void* d_ws, size_t ws_size,
                              hipStream_t stream) {
  const float* bert = (const float*)d_in[0];
  const unsigned char* mask = (const unsigned char*)d_in[1];
  const float* Wih_f = (const float*)d_in[2];
  const float* Whh_f = (const float*)d_in[3];
  const float* bih_f = (const float*)d_in[4];
  const float* bhh_f = (const float*)d_in[5];
  const float* Wih_b = (const float*)d_in[6];
  const float* Whh_b = (const float*)d_in[7];
  const float* bih_b = (const float*)d_in[8];
  const float* bhh_b = (const float*)d_in[9];
  const float* W1 = (const float*)d_in[10];
  const float* b1 = (const float*)d_in[11];
  const float* W2 = (const float*)d_in[12];
  const float* b2 = (const float*)d_in[13];
  const float* Wc = (const float*)d_in[14];
  const float* bc = (const float*)d_in[15];
  const float* start = (const float*)d_in[16];
  const float* endt = (const float*)d_in[17];
  const float* trans = (const float*)d_in[18];
  int* out = (int*)d_out;

  char* base = (char*)d_ws;
  size_t off = 0;
  auto take = [&](size_t bytes) {
    char* p = base + off;
    off = (off + bytes + 255) & ~(size_t)255;
    return p;
  };
  u16* xb = (u16*)take((size_t)MROWS * DD * 2);
  u16* wihp = (u16*)take((size_t)2 * 24 * 96 * 512 * 2);
  u16* whhp = (u16*)take((size_t)2 * 12 * 96 * 512 * 2);
  float* bias = (float*)take((size_t)2 * G4 * 4);
  float* xg = (float*)take((size_t)2 * MROWS * G4 * 4);
  u16* hst = (u16*)take((size_t)2 * 2 * BB * HH * 2);
  u16* hcat = (u16*)take((size_t)MROWS * 2 * HH * 2);
  float* tmp1 = (float*)take((size_t)128 * 768 * 4);
  float* weff = (float*)take((size_t)TT * 768 * 4);
  float* beff = (float*)take(256);
  float* em = (float*)take((size_t)MROWS * TT * 4);
  int* hist = (int*)take((size_t)BB * (SS - 1) * TT * 4);
  int* bars = (int*)take(256);

  hipMemsetAsync(hst, 0, (size_t)2 * 2 * BB * HH * 2, stream);
  hipMemsetAsync(bars, 0, 256, stream);

  int ncvt = MROWS * DD;
  k_cvt<<<(ncvt + 255) / 256, 256, 0, stream>>>(bert, xb, ncvt);

  // pack weights into WMMA B-fragment order (bf16)
  k_pack<<<(G4 * DD + 255) / 256, 256, 0, stream>>>(Wih_f, wihp, DD, 96, G4 * DD);
  k_pack<<<(G4 * DD + 255) / 256, 256, 0, stream>>>(Wih_b, wihp + (size_t)24 * 96 * 512, DD, 96, G4 * DD);
  k_pack<<<(G4 * HH + 255) / 256, 256, 0, stream>>>(Whh_f, whhp, HH, 96, G4 * HH);
  k_pack<<<(G4 * HH + 255) / 256, 256, 0, stream>>>(Whh_b, whhp + (size_t)12 * 96 * 512, HH, 96, G4 * HH);
  k_bias<<<(2 * G4 + 255) / 256, 256, 0, stream>>>(bih_f, bhh_f, bih_b, bhh_b, bias);

  // big WMMA GEMM: xg = x @ Wih^T + bias, both directions
  k_gemm_xg<<<2 * 24 * 256, 256, 0, stream>>>(xb, wihp, bias, xg);

  // collapse linear head
  k_tmp1<<<(128 * 768 + 255) / 256, 256, 0, stream>>>(W2, W1, tmp1);
  k_weff<<<(TT * 768 + 255) / 256, 256, 0, stream>>>(Wc, tmp1, weff);
  k_beff<<<1, 128, 0, stream>>>(W2, b1, b2, Wc, bc, beff);

  // persistent bidirectional LSTM (24 WGs/direction, device barrier per step)
  k_lstm<<<48, 128, 0, stream>>>(xg, whhp, hst, hcat, bars, bars + 2);

  // emissions via collapsed head
  k_emis<<<(MROWS * TT + 255) / 256, 256, 0, stream>>>(hcat, weff, beff, em);

  // CRF Viterbi decode
  k_vit<<<BB, 32, 0, stream>>>(em, mask, start, endt, trans, hist, out);
}
